// DecoderModel_4217657884693
// MI455X (gfx1250) — matrix operand order repaired and checked by reading, not compiled, not run on hardware
//
#include <hip/hip_runtime.h>
#include <cstdint>
#include <cstddef>

// ---------------------------------------------------------------------------
// MI455X / gfx1250 decoder forward.  Matmuls via v_wmma_f32_16x16x32_bf16;
// A-operand tiles staged global->LDS via the Tensor Data Mover (TDM).
// ---------------------------------------------------------------------------

#define DEV_INLINE __device__ __forceinline__

typedef __bf16 bf16_t;
typedef bf16_t v16bf __attribute__((ext_vector_type(16)));
typedef bf16_t v8bf  __attribute__((ext_vector_type(8)));
typedef float  v8f   __attribute__((ext_vector_type(8)));
typedef unsigned int u32x4 __attribute__((ext_vector_type(4)));
typedef int          i32x8 __attribute__((ext_vector_type(8)));
typedef int          i32x4 __attribute__((ext_vector_type(4)));

union BF16Frag { v16bf v; v8bf h[2]; };

// Model constants (match reference)
constexpr int   Bc   = 2;
constexpr int   Sc   = 1024;
constexpr int   Hc   = 2048;
constexpr int   NHc  = 32;
constexpr int   NKVc = 8;
constexpr int   HDc  = 64;
constexpr int   FFc  = 8192;
constexpr int   Lc   = 2;
constexpr int   Mc   = Bc * Sc;          // 2048 token rows
constexpr float EPSc = 1e-5f;
constexpr float NEGc = -65504.0f;        // fp16 min, as in reference mask

DEV_INLINE v8f zero_v8f() {
    v8f z;
#pragma unroll
    for (int i = 0; i < 8; ++i) z[i] = 0.0f;
    return z;
}

DEV_INLINE v8f wmma_bf16(const BF16Frag& a, const BF16Frag& b, v8f c) {
    // 8 args: (neg_a, A, neg_b, B, c_mod, C, reuse_a, reuse_b)
    return __builtin_amdgcn_wmma_f32_16x16x32_bf16(false, a.v, false, b.v,
                                                   (short)0, c, false, false);
}

// ---------------------------------------------------------------------------
// TDM: DMA a [rows x 32] bf16 tile (row stride = kdim elements) from global
// into LDS.  D# per CDNA5 ISA ch.8: group0 {count, lds_addr, global_addr,
// type=2}, group1 {data_size=2B, tensor dims/strides, tile dims}, groups 2/3
// zero (tensor <= 2D).  Issue from one wave; EXEC is ignored by TDM.
// This toolchain's builtin is the 6-arg form:
//   (u32x4 g0, i32x8 g1, i32x4 g2, i32x4 g3, i32x8 pad, i32 cpol)
// ---------------------------------------------------------------------------
DEV_INLINE void tdm_load_tile_bf16(const bf16_t* gp, const bf16_t* lp,
                                   unsigned kdim, unsigned rows)
{
    const unsigned long long ga = (unsigned long long)(uintptr_t)gp;
    const unsigned lds = (unsigned)(uintptr_t)lp;   // flat LDS addr low 32b = offset
    u32x4 g0;
    g0[0] = 1u;                                  // count=1, user descriptor
    g0[1] = lds;                                 // lds_addr
    g0[2] = (unsigned)ga;                        // global_addr[31:0]
    g0[3] = (unsigned)(ga >> 32) | (2u << 30);   // global_addr[56:32] | type=2
    i32x8 g1;
    g1[0] = (int)(1u << 16);                     // data_size=1 -> 2 bytes
    g1[1] = (int)((kdim & 0xFFFFu) << 16);       // tensor_dim0[15:0]
    g1[2] = (int)((kdim >> 16) | ((rows & 0xFFFFu) << 16)); // dim0 hi | dim1 lo
    g1[3] = (int)((rows >> 16) | (32u << 16));   // dim1 hi | tile_dim0=32
    g1[4] = (int)rows;                           // tile_dim1 (tile_dim2=0)
    g1[5] = (int)kdim;                           // tensor_dim0_stride[31:0]
    g1[6] = 0;                                   // stride0 hi | stride1 lo
    g1[7] = 0;
    i32x4 z4;
    z4[0] = 0; z4[1] = 0; z4[2] = 0; z4[3] = 0;
    i32x8 z8;
#pragma unroll
    for (int i = 0; i < 8; ++i) z8[i] = 0;
    __builtin_amdgcn_tensor_load_to_lds(g0, g1, z4, z4, z8, 0);
}

// ---------------------------------------------------------------------------
// Embedding gather: x[row,:] = embed[tokens[row],:]
// ---------------------------------------------------------------------------
__global__ __launch_bounds__(256)
void embed_kernel(const int* __restrict__ tokens,
                  const float* __restrict__ embed,
                  float* __restrict__ x)
{
    const int row = blockIdx.x;
    const int tok = tokens[row];
    const float4* src = (const float4*)(embed + (size_t)tok * Hc);
    float4* dst = (float4*)(x + (size_t)row * Hc);
    for (int i = threadIdx.x; i < Hc / 4; i += 256) dst[i] = src[i];
}

// ---------------------------------------------------------------------------
// RMSNorm: out = w * x * rsqrt(mean(x^2)+eps).
// ---------------------------------------------------------------------------
template <typename OutT>
__global__ __launch_bounds__(256)
void rmsnorm_kernel(const float* __restrict__ x, const float* __restrict__ w,
                    OutT* __restrict__ out, int Hdim)
{
    const int row = blockIdx.x;
    const float* xr = x + (size_t)row * Hdim;
    float ss = 0.0f;
    for (int i = threadIdx.x; i < Hdim; i += 256) { float v = xr[i]; ss += v * v; }
#pragma unroll
    for (int off = 16; off >= 1; off >>= 1) ss += __shfl_xor(ss, off, 32);

    __shared__ float red[8];
    __shared__ float tot;
    const int wid = threadIdx.x >> 5, lane = threadIdx.x & 31;
    if (lane == 0) red[wid] = ss;
    __syncthreads();
    if (threadIdx.x == 0) {
        float s = 0.0f;
#pragma unroll
        for (int i = 0; i < 8; ++i) s += red[i];
        tot = rsqrtf(s / (float)Hdim + EPSc);
    }
    __syncthreads();
    const float r = tot;
    OutT* orow = out + (size_t)row * Hdim;
    for (int i = threadIdx.x; i < Hdim; i += 256)
        orow[i] = (OutT)(w[i] * xr[i] * r);
}

// ---------------------------------------------------------------------------
// GEMM: C[M,N](f32) = A[M,K](bf16) @ W[K,N](f32 -> bf16 in LDS).
// Block 128x128x32, 8 waves 4(m)x2(n), wave tile 32x64 (2x4 WMMA tiles).
// A tile staged by TDM (wave 0), W tile converted per-thread.  Double buffer.
// EPI==0: C = acc        EPI==1: C = resid + acc
// ---------------------------------------------------------------------------
template <int EPI>
__global__ __launch_bounds__(256)
void gemm_bf16_kernel(const bf16_t* __restrict__ A, const float* __restrict__ W,
                      const float* __restrict__ resid, float* __restrict__ C,
                      int Ndim, int Kdim)
{
    constexpr int BM = 128, BN = 128, BK = 32;
    __shared__ __align__(16) bf16_t As[2][BM * BK];   // [m][k] row-major
    __shared__ __align__(16) bf16_t Bs[2][BN * BK];   // [n][k] column-major

    const int t    = threadIdx.x;
    const int bm   = blockIdx.y * BM;
    const int bn   = blockIdx.x * BN;
    const int wv   = t >> 5;
    const int lane = t & 31;
    const int hgrp = lane >> 4;       // lane half-group (0 or 1)
    const int l16  = lane & 15;
    const int wm   = (wv >> 1) * 32;  // wave row offset in tile
    const int wn   = (wv & 1) * 64;   // wave col offset in tile

    v8f acc[2][4];
#pragma unroll
    for (int i = 0; i < 2; ++i)
#pragma unroll
        for (int j = 0; j < 4; ++j) acc[i][j] = zero_v8f();

    auto loadB = [&](int k0, int buf) {
        if (k0 + BK < Kdim)  // speculative prefetch of next W strip
            __builtin_prefetch(W + (size_t)(k0 + BK) * Ndim + bn, 0, 1);
#pragma unroll
        for (int i = 0; i < 4; ++i) {
            int c = t + i * 256;          // 1024 chunks of 4 floats
            int k = c >> 5, nseg = c & 31;
            const float4 f =
                *(const float4*)(W + (size_t)(k0 + k) * Ndim + bn + nseg * 4);
            int nb = nseg * 4;
            Bs[buf][(nb + 0) * BK + k] = (bf16_t)f.x;
            Bs[buf][(nb + 1) * BK + k] = (bf16_t)f.y;
            Bs[buf][(nb + 2) * BK + k] = (bf16_t)f.z;
            Bs[buf][(nb + 3) * BK + k] = (bf16_t)f.w;
        }
    };

    // Prologue: TDM copies A tile 0, threads convert W tile 0.
    if (wv == 0)
        tdm_load_tile_bf16(A + (size_t)bm * Kdim, &As[0][0], (unsigned)Kdim, BM);
    loadB(0, 0);
    if (wv == 0) __builtin_amdgcn_s_wait_tensorcnt(0);
    __syncthreads();

    const int nk = Kdim / BK;
    for (int kt = 0; kt < nk; ++kt) {
        const int buf = kt & 1;
        if (kt + 1 < nk) {
            if (wv == 0)
                tdm_load_tile_bf16(A + (size_t)bm * Kdim + (kt + 1) * BK,
                                   &As[buf ^ 1][0], (unsigned)Kdim, BM);
            loadB((kt + 1) * BK, buf ^ 1);
        }

        BF16Frag af[2], bfr[4];
#pragma unroll
        for (int mt = 0; mt < 2; ++mt) {
            const bf16_t* p = &As[buf][(wm + mt * 16 + l16) * BK];
            af[mt].h[0] = *(const v8bf*)(p + hgrp * 16);         // K {0..7}/{16..23}
            af[mt].h[1] = *(const v8bf*)(p + hgrp * 16 + 8);     // K {8..15}/{24..31}
        }
#pragma unroll
        for (int nt = 0; nt < 4; ++nt) {
            const bf16_t* p = &Bs[buf][(wn + nt * 16 + l16) * BK];
            bfr[nt].h[0] = *(const v8bf*)(p + hgrp * 16);        // K {0..15}/{16..31}
            bfr[nt].h[1] = *(const v8bf*)(p + hgrp * 16 + 8);
        }
#pragma unroll
        for (int mt = 0; mt < 2; ++mt)
#pragma unroll
            for (int nt = 0; nt < 4; ++nt)
                acc[mt][nt] = wmma_bf16(af[mt], bfr[nt], acc[mt][nt]);

        if (wv == 0) __builtin_amdgcn_s_wait_tensorcnt(0);
        __syncthreads();
    }

    // Epilogue: C layout -> row M = r + 8*hgrp, col N = l16 within each tile
#pragma unroll
    for (int mt = 0; mt < 2; ++mt)
#pragma unroll
        for (int nt = 0; nt < 4; ++nt)
#pragma unroll
            for (int r = 0; r < 8; ++r) {
                int gm = bm + wm + mt * 16 + hgrp * 8 + r;
                int gn = bn + wn + nt * 16 + l16;
                size_t idx = (size_t)gm * Ndim + gn;
                float v = acc[mt][nt][r];
                if constexpr (EPI == 1) v += resid[idx];
                C[idx] = v;
            }
}

// ---------------------------------------------------------------------------
// Dual GEMM + SwiGLU: out(bf16) = silu(A@Wg) * (A@Wu).  Block 128x64x32.
// ---------------------------------------------------------------------------
__global__ __launch_bounds__(256)
void gemm_swiglu_kernel(const bf16_t* __restrict__ A, const float* __restrict__ Wgp,
                        const float* __restrict__ Wup, bf16_t* __restrict__ Out,
                        int Ndim, int Kdim)
{
    constexpr int BM = 128, BN = 64, BK = 32;
    __shared__ __align__(16) bf16_t As[2][BM * BK];
    __shared__ __align__(16) bf16_t Bg[2][BN * BK];
    __shared__ __align__(16) bf16_t Bu[2][BN * BK];

    const int t    = threadIdx.x;
    const int bm   = blockIdx.y * BM;
    const int bn   = blockIdx.x * BN;
    const int wv   = t >> 5;
    const int lane = t & 31;
    const int hgrp = lane >> 4;
    const int l16  = lane & 15;
    const int wm   = (wv >> 1) * 32;
    const int wn   = (wv & 1) * 32;

    v8f ag[2][2], au[2][2];
#pragma unroll
    for (int i = 0; i < 2; ++i)
#pragma unroll
        for (int j = 0; j < 2; ++j) { ag[i][j] = zero_v8f(); au[i][j] = zero_v8f(); }

    auto loadB = [&](int k0, int buf) {
        if (k0 + BK < Kdim) {
            __builtin_prefetch(Wgp + (size_t)(k0 + BK) * Ndim + bn, 0, 1);
            __builtin_prefetch(Wup + (size_t)(k0 + BK) * Ndim + bn, 0, 1);
        }
#pragma unroll
        for (int i = 0; i < 2; ++i) {
            int c = t + i * 256;          // 512 chunks of 4 floats per matrix
            int k = c >> 4, nseg = c & 15;
            const float4 fg =
                *(const float4*)(Wgp + (size_t)(k0 + k) * Ndim + bn + nseg * 4);
            const float4 fu =
                *(const float4*)(Wup + (size_t)(k0 + k) * Ndim + bn + nseg * 4);
            int nb = nseg * 4;
            Bg[buf][(nb + 0) * BK + k] = (bf16_t)fg.x;
            Bg[buf][(nb + 1) * BK + k] = (bf16_t)fg.y;
            Bg[buf][(nb + 2) * BK + k] = (bf16_t)fg.z;
            Bg[buf][(nb + 3) * BK + k] = (bf16_t)fg.w;
            Bu[buf][(nb + 0) * BK + k] = (bf16_t)fu.x;
            Bu[buf][(nb + 1) * BK + k] = (bf16_t)fu.y;
            Bu[buf][(nb + 2) * BK + k] = (bf16_t)fu.z;
            Bu[buf][(nb + 3) * BK + k] = (bf16_t)fu.w;
        }
    };

    if (wv == 0)
        tdm_load_tile_bf16(A + (size_t)bm * Kdim, &As[0][0], (unsigned)Kdim, BM);
    loadB(0, 0);
    if (wv == 0) __builtin_amdgcn_s_wait_tensorcnt(0);
    __syncthreads();

    const int nk = Kdim / BK;
    for (int kt = 0; kt < nk; ++kt) {
        const int buf = kt & 1;
        if (kt + 1 < nk) {
            if (wv == 0)
                tdm_load_tile_bf16(A + (size_t)bm * Kdim + (kt + 1) * BK,
                                   &As[buf ^ 1][0], (unsigned)Kdim, BM);
            loadB((kt + 1) * BK, buf ^ 1);
        }

        BF16Frag af[2], bg[2], bu[2];
#pragma unroll
        for (int mt = 0; mt < 2; ++mt) {
            const bf16_t* p = &As[buf][(wm + mt * 16 + l16) * BK];
            af[mt].h[0] = *(const v8bf*)(p + hgrp * 16);
            af[mt].h[1] = *(const v8bf*)(p + hgrp * 16 + 8);
        }
#pragma unroll
        for (int nt = 0; nt < 2; ++nt) {
            const bf16_t* pg = &Bg[buf][(wn + nt * 16 + l16) * BK];
            bg[nt].h[0] = *(const v8bf*)(pg + hgrp * 16);
            bg[nt].h[1] = *(const v8bf*)(pg + hgrp * 16 + 8);
            const bf16_t* pu = &Bu[buf][(wn + nt * 16 + l16) * BK];
            bu[nt].h[0] = *(const v8bf*)(pu + hgrp * 16);
            bu[nt].h[1] = *(const v8bf*)(pu + hgrp * 16 + 8);
        }
#pragma unroll
        for (int mt = 0; mt < 2; ++mt)
#pragma unroll
            for (int nt = 0; nt < 2; ++nt) {
                ag[mt][nt] = wmma_bf16(af[mt], bg[nt], ag[mt][nt]);
                au[mt][nt] = wmma_bf16(af[mt], bu[nt], au[mt][nt]);
            }

        if (wv == 0) __builtin_amdgcn_s_wait_tensorcnt(0);
        __syncthreads();
    }

#pragma unroll
    for (int mt = 0; mt < 2; ++mt)
#pragma unroll
        for (int nt = 0; nt < 2; ++nt)
#pragma unroll
            for (int r = 0; r < 8; ++r) {
                int gm = bm + wm + mt * 16 + hgrp * 8 + r;
                int gn = bn + wn + nt * 16 + l16;
                size_t idx = (size_t)gm * Ndim + gn;
                float g = ag[mt][nt][r];
                float u = au[mt][nt][r];
                float s = g * (1.0f / (1.0f + __expf(-g))) * u;  // silu(g)*u
                Out[idx] = (bf16_t)s;
            }
}

// ---------------------------------------------------------------------------
// RoPE + layout shuffle.
// ---------------------------------------------------------------------------
__global__ __launch_bounds__(256)
void rope_kernel(const float* __restrict__ q, const float* __restrict__ k,
                 const float* __restrict__ v, const int* __restrict__ pos_ids,
                 bf16_t* __restrict__ qbf, bf16_t* __restrict__ kbf,
                 bf16_t* __restrict__ vbf, float* __restrict__ kout,
                 float* __restrict__ vout, int layer)
{
    const int row = blockIdx.x;         // b*S + s
    const int b = row / Sc, s = row % Sc;
    const int t = threadIdx.x;
    const float p = (float)pos_ids[row];

#pragma unroll
    for (int i = 0; i < 4; ++i) {
        int pr = t + i * 256;
        int hh = pr >> 5, d = pr & 31;
        float freq = p * powf(10000.0f, -(float)(2 * d) * (1.0f / 64.0f));
        float c = cosf(freq), sn = sinf(freq);
        size_t base = (size_t)row * (NHc * HDc) + (size_t)hh * HDc;
        float x0 = q[base + d], x1 = q[base + d + 32];
        qbf[base + d]      = (bf16_t)(x0 * c - x1 * sn);
        qbf[base + d + 32] = (bf16_t)(x1 * c + x0 * sn);
    }
    {
        int hh = t >> 5, d = t & 31;
        float freq = p * powf(10000.0f, -(float)(2 * d) * (1.0f / 64.0f));
        float c = cosf(freq), sn = sinf(freq);
        size_t base = (size_t)row * (NKVc * HDc) + (size_t)hh * HDc;
        float x0 = k[base + d], x1 = k[base + d + 32];
        float r0 = x0 * c - x1 * sn;
        float r1 = x1 * c + x0 * sn;
        size_t ko = ((size_t)((layer * Bc + b) * NKVc + hh) * Sc + s) * HDc;
        kout[ko + d] = r0; kout[ko + d + 32] = r1;
        size_t kb = ((size_t)(b * NKVc + hh) * Sc + s) * HDc;
        kbf[kb + d] = (bf16_t)r0; kbf[kb + d + 32] = (bf16_t)r1;
    }
#pragma unroll
    for (int i = 0; i < 2; ++i) {
        int e = t + i * 256;
        int hh = e >> 6, d = e & 63;
        float vv = v[(size_t)row * (NKVc * HDc) + (size_t)hh * HDc + d];
        size_t vo = ((size_t)((layer * Bc + b) * NKVc + hh) * Sc + s) * HDc + d;
        vout[vo] = vv;
        size_t vb = ((size_t)(b * NKVc + hh) * Sc + s) * HDc + d;
        vbf[vb] = (bf16_t)vv;
    }
}

// ---------------------------------------------------------------------------
// Flash attention, causal, GQA.  Grid (S/64, B*NH), 128 threads = 4 waves.
// ---------------------------------------------------------------------------
__global__ __launch_bounds__(128)
void attention_kernel(const bf16_t* __restrict__ qbf, const bf16_t* __restrict__ kbf,
                      const bf16_t* __restrict__ vbf, const int* __restrict__ amask,
                      bf16_t* __restrict__ attnout)
{
    __shared__ __align__(16) bf16_t Ks[64 * 64];      // [kv][hd]
    __shared__ __align__(16) bf16_t Vt[64 * 64];      // [hd][kv] (transposed)
    __shared__ __align__(16) bf16_t Pw[4][16 * 64];   // per-wave P tile

    const int t    = threadIdx.x;
    const int wv   = t >> 5;
    const int lane = t & 31;
    const int hgrp = lane >> 4;
    const int l16  = lane & 15;

    const int qblk = blockIdx.x;                      // 0..S/64-1
    const int bh   = blockIdx.y;
    const int b    = bh / NHc;
    const int h    = bh % NHc;
    const int kvh  = h / (NHc / NKVc);
    const int qbase = qblk * 64 + wv * 16;

    BF16Frag qf[2];
    {
        const int sg = qbase + l16;
        const bf16_t* qp = qbf + ((size_t)(b * Sc + sg) * NHc + h) * HDc;
#pragma unroll
        for (int kt = 0; kt < 2; ++kt) {
            qf[kt].h[0] = *(const v8bf*)(qp + kt * 32 + hgrp * 16);
            qf[kt].h[1] = *(const v8bf*)(qp + kt * 32 + hgrp * 16 + 8);
        }
    }

    float mrun[8], lrun[8];
    v8f accO[4];
#pragma unroll
    for (int r = 0; r < 8; ++r) { mrun[r] = -3.0e38f; lrun[r] = 0.0f; }
#pragma unroll
    for (int nt = 0; nt < 4; ++nt) accO[nt] = zero_v8f();

    const bf16_t* kg = kbf + (size_t)(b * NKVc + kvh) * Sc * HDc;
    const bf16_t* vg = vbf + (size_t)(b * NKVc + kvh) * Sc * HDc;
    const float scale = 0.125f;                       // 1/sqrt(64)

    const int ntiles = qblk + 1;                      // causal: kv <= q block
    for (int it = 0; it < ntiles; ++it) {
        const int kb = it * 64;
#pragma unroll
        for (int i = 0; i < 4; ++i) {
            int c = t + i * 128;                      // 512 chunks of 8 bf16
            int kvr = c >> 3, seg = c & 7;
            *(uint4*)&Ks[kvr * 64 + seg * 8] =
                *(const uint4*)(kg + (size_t)(kb + kvr) * HDc + seg * 8);
            v8bf vv = *(const v8bf*)(vg + (size_t)(kb + kvr) * HDc + seg * 8);
#pragma unroll
            for (int j = 0; j < 8; ++j) Vt[(seg * 8 + j) * 64 + kvr] = vv[j];
        }
        __syncthreads();

        v8f sc[4];
#pragma unroll
        for (int jt = 0; jt < 4; ++jt) sc[jt] = zero_v8f();
#pragma unroll
        for (int jt = 0; jt < 4; ++jt)
#pragma unroll
            for (int kt = 0; kt < 2; ++kt) {
                BF16Frag bf_;
                const bf16_t* p = &Ks[(jt * 16 + l16) * 64 + kt * 32 + hgrp * 16];
                bf_.h[0] = *(const v8bf*)(p);
                bf_.h[1] = *(const v8bf*)(p + 8);
                sc[jt] = wmma_bf16(qf[kt], bf_, sc[jt]);
            }

        float pv[4][8];
        float rmax[8];
#pragma unroll
        for (int r = 0; r < 8; ++r) rmax[r] = -3.0e38f;
#pragma unroll
        for (int jt = 0; jt < 4; ++jt) {
            const int kvg = kb + jt * 16 + l16;
            const float padneg = amask[b * Sc + kvg] ? 0.0f : NEGc;
#pragma unroll
            for (int r = 0; r < 8; ++r) {
                const int qg = qbase + hgrp * 8 + r;
                float sval = sc[jt][r] * scale + padneg;
                if (kvg > qg) sval += NEGc;           // causal
                pv[jt][r] = sval;
                rmax[r] = fmaxf(rmax[r], sval);
            }
        }
#pragma unroll
        for (int r = 0; r < 8; ++r)
#pragma unroll
            for (int off = 1; off < 16; off <<= 1)
                rmax[r] = fmaxf(rmax[r], __shfl_xor(rmax[r], off, 32));

#pragma unroll
        for (int r = 0; r < 8; ++r) {
            float mnew = fmaxf(mrun[r], rmax[r]);
            float alpha = __expf(mrun[r] - mnew);
            mrun[r] = mnew;
            lrun[r] *= alpha;
#pragma unroll
            for (int nt = 0; nt < 4; ++nt) accO[nt][r] *= alpha;
        }

        float rsum[8];
#pragma unroll
        for (int r = 0; r < 8; ++r) rsum[r] = 0.0f;
#pragma unroll
        for (int jt = 0; jt < 4; ++jt)
#pragma unroll
            for (int r = 0; r < 8; ++r) {
                float pe = __expf(pv[jt][r] - mrun[r]);
                rsum[r] += pe;
                Pw[wv][(hgrp * 8 + r) * 64 + jt * 16 + l16] = (bf16_t)pe;
            }
#pragma unroll
        for (int r = 0; r < 8; ++r) {
#pragma unroll
            for (int off = 1; off < 16; off <<= 1)
                rsum[r] += __shfl_xor(rsum[r], off, 32);
            lrun[r] += rsum[r];
        }

#pragma unroll
        for (int kt = 0; kt < 2; ++kt) {
            BF16Frag pf;
            const bf16_t* pp = &Pw[wv][l16 * 64 + kt * 32];
            pf.h[0] = *(const v8bf*)(pp + hgrp * 16);
            pf.h[1] = *(const v8bf*)(pp + hgrp * 16 + 8);
#pragma unroll
            for (int nt = 0; nt < 4; ++nt) {
                BF16Frag vf;
                const bf16_t* vp = &Vt[(nt * 16 + l16) * 64 + kt * 32 + hgrp * 16];
                vf.h[0] = *(const v8bf*)(vp);
                vf.h[1] = *(const v8bf*)(vp + 8);
                accO[nt] = wmma_bf16(pf, vf, accO[nt]);
            }
        }
        __syncthreads();
    }

#pragma unroll
    for (int r = 0; r < 8; ++r) {
        const int qg = qbase + hgrp * 8 + r;
        const float inv = 1.0f / lrun[r];
#pragma unroll
        for (int nt = 0; nt < 4; ++nt) {
            float o = accO[nt][r] * inv;
            attnout[((size_t)(b * Sc + qg) * NHc + h) * HDc + nt * 16 + l16] =
                (bf16_t)o;
        }
    }
}

// ---------------------------------------------------------------------------
// Host-side orchestration
// ---------------------------------------------------------------------------
extern "C" void kernel_launch(void* const* d_in, const int* in_sizes, int n_in,
                              void* d_out, int out_size, void* d_ws, size_t ws_size,
                              hipStream_t stream)
{
    (void)in_sizes; (void)n_in; (void)out_size; (void)ws_size;

    const int*   tokens = (const int*)d_in[0];
    const int*   pos    = (const int*)d_in[1];
    const int*   amask  = (const int*)d_in[2];
    // d_in[3..5]: use_cache / empty caches — unused (prefill path)
    const float* embed  = (const float*)d_in[6];
    const float* ln1    = (const float*)d_in[7];
    const float* ln2    = (const float*)d_in[8];
    const float* Wq     = (const float*)d_in[9];
    const float* Wk     = (const float*)d_in[10];
    const float* Wv     = (const float*)d_in[11];
    const float* Wo     = (const float*)d_in[12];
    const float* Wg     = (const float*)d_in[13];
    const float* Wu     = (const float*)d_in[14];
    const float* Wd     = (const float*)d_in[15];
    const float* normw  = (const float*)d_in[16];

    float* out   = (float*)d_out;
    float* x_out = out;
    float* k_out = out + (size_t)Mc * Hc;
    float* v_out = k_out + (size_t)Lc * Bc * NKVc * Sc * HDc;

    char* ws = (char*)d_ws;
    auto carve = [&](size_t bytes) -> char* {
        char* p = ws;
        ws += (bytes + 255) & ~(size_t)255;
        return p;
    };
    float*  x   = (float*)carve((size_t)Mc * Hc * 4);            // residual stream
    float*  qf  = (float*)carve((size_t)Mc * Hc * 4);            // q pre-rope
    float*  kf  = (float*)carve((size_t)Mc * NKVc * HDc * 4);    // k pre-rope
    float*  vf  = (float*)carve((size_t)Mc * NKVc * HDc * 4);    // v
    bf16_t* hbf = (bf16_t*)carve((size_t)Mc * Hc * 2);           // rmsnorm out
    bf16_t* qbf = (bf16_t*)carve((size_t)Mc * Hc * 2);
    bf16_t* kbf = (bf16_t*)carve((size_t)Mc * NKVc * HDc * 2);
    bf16_t* vbf = (bf16_t*)carve((size_t)Mc * NKVc * HDc * 2);
    bf16_t* abf = (bf16_t*)carve((size_t)Mc * Hc * 2);           // attn out
    bf16_t* mbf = (bf16_t*)carve((size_t)Mc * FFc * 2);          // swiglu mid

    embed_kernel<<<Mc, 256, 0, stream>>>(tokens, embed, x);

    for (int l = 0; l < Lc; ++l) {
        rmsnorm_kernel<bf16_t><<<Mc, 256, 0, stream>>>(x, ln1 + (size_t)l * Hc, hbf, Hc);

        gemm_bf16_kernel<0><<<dim3(Hc / 128, Mc / 128), 256, 0, stream>>>(
            hbf, Wq + (size_t)l * Hc * Hc, nullptr, qf, Hc, Hc);
        gemm_bf16_kernel<0><<<dim3((NKVc * HDc) / 128, Mc / 128), 256, 0, stream>>>(
            hbf, Wk + (size_t)l * Hc * NKVc * HDc, nullptr, kf, NKVc * HDc, Hc);
        gemm_bf16_kernel<0><<<dim3((NKVc * HDc) / 128, Mc / 128), 256, 0, stream>>>(
            hbf, Wv + (size_t)l * Hc * NKVc * HDc, nullptr, vf, NKVc * HDc, Hc);

        rope_kernel<<<Mc, 256, 0, stream>>>(qf, kf, vf, pos, qbf, kbf, vbf,
                                            k_out, v_out, l);

        attention_kernel<<<dim3(Sc / 64, Bc * NHc), 128, 0, stream>>>(
            qbf, kbf, vbf, amask, abf);

        gemm_bf16_kernel<1><<<dim3(Hc / 128, Mc / 128), 256, 0, stream>>>(
            abf, Wo + (size_t)l * Hc * Hc, x, x, Hc, Hc);

        rmsnorm_kernel<bf16_t><<<Mc, 256, 0, stream>>>(x, ln2 + (size_t)l * Hc, hbf, Hc);

        gemm_swiglu_kernel<<<dim3(FFc / 64, Mc / 128), 256, 0, stream>>>(
            hbf, Wg + (size_t)l * Hc * FFc, Wu + (size_t)l * Hc * FFc, mbf, FFc, Hc);

        gemm_bf16_kernel<1><<<dim3(Hc / 128, Mc / 128), 256, 0, stream>>>(
            mbf, Wd + (size_t)l * FFc * Hc, x, x, Hc, FFc);
    }

    rmsnorm_kernel<float><<<Mc, 256, 0, stream>>>(x, normw, x_out, Hc);
}
